// GraphSAGE_420906795209
// MI455X (gfx1250) — matrix operand order repaired and checked by reading, not compiled, not run on hardware
//
#include <hip/hip_runtime.h>

typedef __attribute__((ext_vector_type(16))) __bf16 v16bf;
typedef __attribute__((ext_vector_type(8)))  float  v8f;
typedef unsigned short u16;
typedef unsigned int   u32;

__device__ __forceinline__ u16 f2bf(float f) {
  return __builtin_bit_cast(u16, (__bf16)f);   // native hw convert (RNE)
}
__device__ __forceinline__ float bf2f(u16 h) {
  u32 u = ((u32)h) << 16;
  return __builtin_bit_cast(float, u);
}

union Frag { uint4 q[2]; v16bf v; };

// ---- CDNA5 async global->LDS DMA (ASYNCcnt-tracked), 16B per lane ----
__device__ __forceinline__ void async_g2l_b128(void* lds, const void* g) {
  u32 loff = (u32)(uintptr_t)lds;              // low 32 bits = LDS byte offset
  asm volatile("global_load_async_to_lds_b128 %0, %1, off"
               :: "v"(loff), "v"(g) : "memory");
}
__device__ __forceinline__ void wait_async0() {
  asm volatile("s_wait_asynccnt 0x0" ::: "memory");
}

// ---------------------------------------------------------------------------
// Generic fused GEMM: C = act( A0 @ W0 [+ A1 @ W1] + bias )
// A0: M x 256 (fp32 if A0F32 else bf16), A1: M x 256 bf16 (mean)
// W0t/W1t: weights PRE-TRANSPOSED to N x 256 bf16 (so B staging is verbatim copy)
// Block tile 128(M) x 64(N), 256 threads = 8 waves, wave tile 32x32 (2x2 WMMA).
// ---------------------------------------------------------------------------
template<bool A0F32, bool HASA1, bool RELU, bool OUTF32>
__global__ __launch_bounds__(256)
void gemm_wmma(const void* __restrict__ A0v, const u16* __restrict__ A1,
               const u16* __restrict__ W0t, const u16* __restrict__ W1t,
               const float* __restrict__ bias, void* __restrict__ Cv,
               int M, int N)
{
  __shared__ __align__(16) u16 As[128][40];   // 128 rows x 32 k (+pad)
  __shared__ __align__(16) u16 Bs[64][40];    // 64 n-rows x 32 k (+pad)

  const int tid   = threadIdx.x;
  const int lane  = tid & 31;
  const int wid   = tid >> 5;
  const int waveM = wid >> 1;          // 0..3
  const int waveN = wid & 1;           // 0..1
  const int half  = lane >> 4;
  const int l15   = lane & 15;

  const int rb = blockIdx.x * 128;
  const int nb = blockIdx.y * 64;

  v8f acc[2][2];
  acc[0][0] = {}; acc[0][1] = {}; acc[1][0] = {}; acc[1][1] = {};

  const int ar  = tid >> 1;            // A-tile row 0..127
  const int ac  = (tid & 1) * 16;      // A-tile col group: 0 or 16
  const int bnr = tid >> 2;            // B-tile n-row 0..63
  const int bkg = (tid & 3) * 8;       // B-tile k group 0,8,16,24

  const int nparts = HASA1 ? 2 : 1;
  for (int part = 0; part < nparts; ++part) {
    const u16* Wt = (part == 0) ? W0t : W1t;
    for (int kt = 0; kt < 256; kt += 32) {
      __syncthreads();
      // ---- stage A tile ----
      if (A0F32 && part == 0) {
        const float* Ap = (const float*)A0v;
        const float4* s4 = (const float4*)&Ap[(size_t)(rb + ar) * 256 + kt + ac];
        __align__(16) u16 tmp[16];
        #pragma unroll
        for (int i = 0; i < 4; ++i) {
          float4 f = s4[i];
          tmp[i*4+0] = f2bf(f.x); tmp[i*4+1] = f2bf(f.y);
          tmp[i*4+2] = f2bf(f.z); tmp[i*4+3] = f2bf(f.w);
        }
        *(uint4*)&As[ar][ac]     = *(uint4*)&tmp[0];
        *(uint4*)&As[ar][ac + 8] = *(uint4*)&tmp[8];
      } else {
        const u16* Ap = (part == 0) ? (const u16*)A0v : A1;
        const u16* gA = &Ap[(size_t)(rb + ar) * 256 + kt + ac];
        async_g2l_b128(&As[ar][ac],     gA);
        async_g2l_b128(&As[ar][ac + 8], gA + 8);
      }
      // ---- stage B tile (already transposed in memory: Wt[n][k]) ----
      async_g2l_b128(&Bs[bnr][bkg], &Wt[(size_t)(nb + bnr) * 256 + kt + bkg]);
      wait_async0();
      __syncthreads();

      // ---- fragments per CDNA5 VGPR layouts ----
      Frag a[2], b[2];
      #pragma unroll
      for (int mi = 0; mi < 2; ++mi) {
        const int row = waveM * 32 + mi * 16 + l15;
        a[mi].q[0] = *(const uint4*)&As[row][half * 8];        // k = half*8 .. +7
        a[mi].q[1] = *(const uint4*)&As[row][16 + half * 8];   // k = 16+half*8 .. +7
      }
      #pragma unroll
      for (int ni = 0; ni < 2; ++ni) {
        const int col = waveN * 32 + ni * 16 + l15;
        b[ni].q[0] = *(const uint4*)&Bs[col][half * 8];        // k = half*8 .. +7
        b[ni].q[1] = *(const uint4*)&Bs[col][16 + half * 8];   // k = 16+half*8 .. +7
      }
      #pragma unroll
      for (int mi = 0; mi < 2; ++mi)
        #pragma unroll
        for (int ni = 0; ni < 2; ++ni)
          acc[mi][ni] = __builtin_amdgcn_wmma_f32_16x16x32_bf16(
              false, a[mi].v, false, b[ni].v, (short)0, acc[mi][ni], false, false);
    }
  }

  // ---- epilogue: bias + optional ReLU, store bf16 or fp32 ----
  #pragma unroll
  for (int mi = 0; mi < 2; ++mi) {
    #pragma unroll
    for (int ni = 0; ni < 2; ++ni) {
      const int col = nb + waveN * 32 + ni * 16 + l15;
      const float bv = bias[col];
      #pragma unroll
      for (int r = 0; r < 8; ++r) {
        const int row = rb + waveM * 32 + mi * 16 + half * 8 + r;
        float v = acc[mi][ni][r] + bv;
        if (RELU) v = v > 0.0f ? v : 0.0f;
        if (OUTF32) ((float*)Cv)[(size_t)row * N + col] = v;
        else        ((u16*)Cv)[(size_t)row * N + col]   = f2bf(v);
      }
    }
  }
}

// ---------------------------------------------------------------------------
// Edge scatter: agg[dst] += bf16(h[src]) in fp32, cnt[dst] += 1. 32 lanes/edge.
// ---------------------------------------------------------------------------
__global__ __launch_bounds__(256)
void scatter_edges(const u16* __restrict__ h, const int* __restrict__ src,
                   const int* __restrict__ dst, float* __restrict__ agg,
                   float* __restrict__ cnt, int E)
{
  const long long gid = (long long)blockIdx.x * 256 + threadIdx.x;
  const long long e = gid >> 5;
  if (e >= E) return;
  const int l = (int)(gid & 31);
  const int s = src[e];
  const int d = dst[e];
  // prefetch a future edge's source row to hide random-gather latency
  if (l == 0 && e + 256 < E)
    __builtin_prefetch(&h[(size_t)src[e + 256] * 256], 0, 0);
  const uint4 q = *(const uint4*)&h[(size_t)s * 256 + l * 8];
  const u16* hv = (const u16*)&q;
  float* ap = &agg[(size_t)d * 256 + l * 8];
  #pragma unroll
  for (int j = 0; j < 8; ++j) atomicAdd(ap + j, bf2f(hv[j]));
  if (l == 0) atomicAdd(&cnt[d], 1.0f);
}

__global__ __launch_bounds__(256)
void mean_to_bf16(const float* __restrict__ agg, const float* __restrict__ cnt,
                  u16* __restrict__ mean, int n_dst)
{
  const int gid = blockIdx.x * 256 + threadIdx.x;
  if (gid >= n_dst * 64) return;
  const int row = gid >> 6;
  const int c = (gid & 63) * 4;
  float cn = cnt[row];
  cn = cn < 1.0f ? 1.0f : cn;
  const float inv = 1.0f / cn;
  const float4 a = *(const float4*)&agg[(size_t)row * 256 + c];
  __align__(8) u16 o[4] = { f2bf(a.x*inv), f2bf(a.y*inv), f2bf(a.z*inv), f2bf(a.w*inv) };
  *(uint2*)&mean[(size_t)row * 256 + c] = *(uint2*)o;
}

__global__ __launch_bounds__(256)
void zero_f4(float4* __restrict__ p, int n4)
{
  const int gid = blockIdx.x * 256 + threadIdx.x;
  if (gid < n4) p[gid] = make_float4(0.f, 0.f, 0.f, 0.f);
}

// Convert fp32 weight (K=256 x N row-major) to bf16 TRANSPOSED (N x 256).
__global__ __launch_bounds__(256)
void convT_f32_bf16(const float* __restrict__ in, u16* __restrict__ out, int N)
{
  const int gid = blockIdx.x * 256 + threadIdx.x;   // over 256*N
  if (gid >= 256 * N) return;
  const int n = gid >> 8;       // 0..N-1
  const int k = gid & 255;      // 0..255
  out[(size_t)n * 256 + k] = f2bf(in[(size_t)k * N + n]);
}

// ---------------------------------------------------------------------------
extern "C" void kernel_launch(void* const* d_in, const int* in_sizes, int n_in,
                              void* d_out, int out_size, void* d_ws, size_t ws_size,
                              hipStream_t stream)
{
  const float* x    = (const float*)d_in[0];
  const int*   src0 = (const int*)d_in[1];
  const int*   dst0 = (const int*)d_in[2];
  const int*   src1 = (const int*)d_in[3];
  const int*   dst1 = (const int*)d_in[4];
  const int*   src2 = (const int*)d_in[5];
  const int*   dst2 = (const int*)d_in[6];
  const float* Wfc  = (const float*)d_in[10];
  const float* bfc  = (const float*)d_in[11];
  const float* Ws0  = (const float*)d_in[12];
  const float* Wn0  = (const float*)d_in[13];
  const float* b0   = (const float*)d_in[14];
  const float* Ws1  = (const float*)d_in[15];
  const float* Wn1  = (const float*)d_in[16];
  const float* b1   = (const float*)d_in[17];
  const float* Ws2  = (const float*)d_in[18];
  const float* Wn2  = (const float*)d_in[19];
  const float* b2   = (const float*)d_in[20];

  const int E0 = in_sizes[1], E1 = in_sizes[3], E2 = in_sizes[5];
  const int N0 = 524288, N1 = 262144, N2 = 65536, N3 = 8192;

  char* ws = (char*)d_ws;
  size_t off = 0;
  auto carve = [&](size_t bytes) -> char* {
    char* p = ws + off; off += (bytes + 255) & ~(size_t)255; return p;
  };
  u16* wWfc = (u16*)carve((size_t)65536 * 2);
  u16* wWs0 = (u16*)carve((size_t)65536 * 2);
  u16* wWn0 = (u16*)carve((size_t)65536 * 2);
  u16* wWs1 = (u16*)carve((size_t)65536 * 2);
  u16* wWn1 = (u16*)carve((size_t)65536 * 2);
  u16* wWs2 = (u16*)carve((size_t)16384 * 2);
  u16* wWn2 = (u16*)carve((size_t)16384 * 2);
  u16*   h0   = (u16*)carve((size_t)N0 * 256 * 2);
  u16*   h1   = (u16*)carve((size_t)N1 * 256 * 2);
  u16*   mn   = (u16*)carve((size_t)N1 * 256 * 2);
  float* agg  = (float*)carve((size_t)N1 * 256 * 4);
  float* cnt  = (float*)carve((size_t)N1 * 4);
  u16*   h2   = h0;   // h0 dead after layer-0 GEMM; reuse for h2

  // --- weight conversion fp32 -> bf16, transposed to N x 256 ---
  convT_f32_bf16<<<256, 256, 0, stream>>>(Wfc, wWfc, 256);
  convT_f32_bf16<<<256, 256, 0, stream>>>(Ws0, wWs0, 256);
  convT_f32_bf16<<<256, 256, 0, stream>>>(Wn0, wWn0, 256);
  convT_f32_bf16<<<256, 256, 0, stream>>>(Ws1, wWs1, 256);
  convT_f32_bf16<<<256, 256, 0, stream>>>(Wn1, wWn1, 256);
  convT_f32_bf16<<< 64, 256, 0, stream>>>(Ws2, wWs2, 64);
  convT_f32_bf16<<< 64, 256, 0, stream>>>(Wn2, wWn2, 64);

  // --- FC: h0 = relu(x @ Wfc + bfc) ---
  gemm_wmma<true, false, true, false><<<dim3(N0 / 128, 256 / 64), 256, 0, stream>>>(
      x, nullptr, wWfc, nullptr, bfc, h0, N0, 256);

  // --- Layer 0 ---
  zero_f4<<<(N1 * 64 + 255) / 256, 256, 0, stream>>>((float4*)agg, N1 * 64);
  zero_f4<<<(N1 / 4 + 255) / 256, 256, 0, stream>>>((float4*)cnt, N1 / 4);
  scatter_edges<<<(int)(((long long)E0 * 32 + 255) / 256), 256, 0, stream>>>(
      h0, src0, dst0, agg, cnt, E0);
  mean_to_bf16<<<(N1 * 64 + 255) / 256, 256, 0, stream>>>(agg, cnt, mn, N1);
  gemm_wmma<false, true, true, false><<<dim3(N1 / 128, 256 / 64), 256, 0, stream>>>(
      h0, mn, wWs0, wWn0, b0, h1, N1, 256);

  // --- Layer 1 ---
  zero_f4<<<(N2 * 64 + 255) / 256, 256, 0, stream>>>((float4*)agg, N2 * 64);
  zero_f4<<<(N2 / 4 + 255) / 256, 256, 0, stream>>>((float4*)cnt, N2 / 4);
  scatter_edges<<<(int)(((long long)E1 * 32 + 255) / 256), 256, 0, stream>>>(
      h1, src1, dst1, agg, cnt, E1);
  mean_to_bf16<<<(N2 * 64 + 255) / 256, 256, 0, stream>>>(agg, cnt, mn, N2);
  gemm_wmma<false, true, true, false><<<dim3(N2 / 128, 256 / 64), 256, 0, stream>>>(
      h1, mn, wWs1, wWn1, b1, h2, N2, 256);

  // --- Layer 2 (fp32 output, no ReLU) ---
  zero_f4<<<(N3 * 64 + 255) / 256, 256, 0, stream>>>((float4*)agg, N3 * 64);
  zero_f4<<<(N3 / 4 + 255) / 256, 256, 0, stream>>>((float4*)cnt, N3 / 4);
  scatter_edges<<<(int)(((long long)E2 * 32 + 255) / 256), 256, 0, stream>>>(
      h2, src2, dst2, agg, cnt, E2);
  mean_to_bf16<<<(N3 * 64 + 255) / 256, 256, 0, stream>>>(agg, cnt, mn, N3);
  gemm_wmma<false, true, false, true><<<dim3(N3 / 128, 64 / 64), 256, 0, stream>>>(
      h2, mn, wWs2, wWn2, b2, d_out, N3, 64);
}
